// Pointnet2BackboneSeg_53523882443490
// MI455X (gfx1250) — compile-verified, matrix operand order checked by reading, not run on hardware
//
#include <hip/hip_runtime.h>
#include <hip/hip_bf16.h>
#include <stdint.h>

typedef __attribute__((ext_vector_type(16))) _Float16 v16h;
typedef __attribute__((ext_vector_type(8)))  float    v8f;
typedef __attribute__((ext_vector_type(4))) unsigned int u32x4;
typedef __attribute__((ext_vector_type(8))) int i32x8;
typedef __attribute__((ext_vector_type(4))) int i32x4;
typedef _Float16 f16;

// ---------------------------------------------------------------------------
// WMMA 16x16x32 f16 helpers (CDNA5 wave32 layouts, ISA 7.12.2)
//   A (16xK slice): dword q of the v16h for lane:
//     q<4 : K = (lane/16)*8 + 2q          (K 0..15 region)
//     q>=4: K = 16 + (lane/16)*8 + 2(q-4) (K 16..31 region)
//   B (Kx16): mirrored, lane holds column n=lane%16, same K pairing.
//   C/D f32: c[i] -> M = i + 8*(lane/16), N = lane%16.
// ---------------------------------------------------------------------------
__device__ __forceinline__ int kbase_pair(int q, int hid) {
  return (q < 4) ? (hid * 8 + 2 * q) : (16 + hid * 8 + 2 * (q - 4));
}

// One dense MLP layer: out[rows,Cout] = relu(in[rows,Kdim] * W^T + bias)
// in: LDS f16 (ldin stride); W: global f16, row-major (Cout x Kdim, K padded
// to mult of 32); out: LDS f16 (ldout stride). rows,Cout mult of 16.
__device__ __forceinline__ void wmma_mlp_layer(
    const f16* __restrict__ sh_in, int ldin,
    f16* __restrict__ sh_out, int ldout,
    const f16* __restrict__ Wg, const float* __restrict__ bias,
    int rows, int Kdim, int Cout, int tid, int nthreads)
{
  const int lane   = tid & 31;
  const int wave   = tid >> 5;
  const int nwaves = nthreads >> 5;
  const int r      = lane & 15;   // N (and A-row) index within tile
  const int hid    = lane >> 4;   // half-wave id
  const int Mt = rows >> 4, Nt = Cout >> 4, Kt = Kdim >> 5;
  for (int t = wave; t < Mt * Nt; t += nwaves) {
    const int mt = t / Nt;
    const int nt = t - mt * Nt;
    const float bv = bias[nt * 16 + r];
    v8f acc;
#pragma unroll
    for (int i = 0; i < 8; ++i) acc[i] = bv;
    for (int kt = 0; kt < Kt; ++kt) {
      union { uint32_t u[8]; v16h v; } A, Bf;
      const f16* arow = sh_in + (size_t)(mt * 16 + r) * ldin + kt * 32;
      const f16* brow = Wg    + (size_t)(nt * 16 + r) * Kdim + kt * 32;
#pragma unroll
      for (int q = 0; q < 8; ++q) {
        const int kb = kbase_pair(q, hid);
        A.u[q]  = *(const uint32_t*)(arow + kb);
        Bf.u[q] = *(const uint32_t*)(brow + kb);
      }
      acc = __builtin_amdgcn_wmma_f32_16x16x32_f16(false, A.v, false, Bf.v,
                                                   (short)0, acc, false, false);
    }
#pragma unroll
    for (int i = 0; i < 8; ++i) {
      float x = acc[i];
      x = x > 0.f ? x : 0.f;  // relu
      sh_out[(size_t)(mt * 16 + hid * 8 + i) * ldout + nt * 16 + r] = (f16)x;
    }
  }
}

// ---------------------------------------------------------------------------
// Weight conversion: f32 (Cout x Cin) -> f16 (Cout x Cin_pad, zero padded)
// ---------------------------------------------------------------------------
__global__ void k_convert_w(const float* __restrict__ W, f16* __restrict__ dst,
                            int Cout, int Cin, int Cin_pad) {
  int t = blockIdx.x * blockDim.x + threadIdx.x;
  int total = Cout * Cin_pad;
  if (t >= total) return;
  int o = t / Cin_pad, k = t - o * Cin_pad;
  dst[t] = (f16)((k < Cin) ? W[(size_t)o * Cin + k] : 0.f);
}

// ---------------------------------------------------------------------------
// Farthest point sampling: one block per batch, sequential argmax iterations.
// ---------------------------------------------------------------------------
__global__ void k_fps(const float* __restrict__ xyz, int n, int npoint,
                      int* __restrict__ inds, float* __restrict__ dists) {
  const int b = blockIdx.x;
  const int tid = threadIdx.x;
  const float* P = xyz + (size_t)b * n * 3;
  float* D = dists + (size_t)b * n;
  __shared__ float smax[256];
  __shared__ int   simx[256];
  for (int j = tid; j < n; j += 256) D[j] = 1e10f;
  __syncthreads();
  int far = 0;
  for (int it = 0; it < npoint; ++it) {
    if (tid == 0) inds[(size_t)b * npoint + it] = far;
    const float fx = P[far * 3 + 0], fy = P[far * 3 + 1], fz = P[far * 3 + 2];
    float bm = -1.f; int bi = 0;
    for (int j = tid; j < n; j += 256) {
      float dx = P[j * 3 + 0] - fx;
      float dy = P[j * 3 + 1] - fy;
      float dz = P[j * 3 + 2] - fz;
      float dd = dx * dx + dy * dy + dz * dz;
      float od = D[j];
      float nv = dd < od ? dd : od;
      D[j] = nv;
      if (nv > bm) { bm = nv; bi = j; }
    }
    smax[tid] = bm; simx[tid] = bi;
    __syncthreads();
    for (int off = 128; off > 0; off >>= 1) {
      if (tid < off && smax[tid + off] > smax[tid]) {
        smax[tid] = smax[tid + off]; simx[tid] = simx[tid + off];
      }
      __syncthreads();
    }
    far = simx[0];
    __syncthreads();
  }
}

__global__ void k_gather_xyz(const float* __restrict__ src, const int* __restrict__ inds,
                             float* __restrict__ dst, int n_src, int S, int B) {
  int t = blockIdx.x * blockDim.x + threadIdx.x;
  if (t >= B * S) return;
  int b = t / S;
  int id = inds[t];
  const float* p = src + ((size_t)b * n_src + id) * 3;
  dst[t * 3 + 0] = p[0]; dst[t * 3 + 1] = p[1]; dst[t * 3 + 2] = p[2];
}

// ---------------------------------------------------------------------------
// Ball query: one wave32 per query point; ballot-compaction keeps index order
// (matches reference's sort-by-index-then-truncate semantics).
// ---------------------------------------------------------------------------
__global__ void k_ball_query(const float* __restrict__ xyz, const float* __restrict__ nxyz,
                             int n, int S, int ns, float r2, int* __restrict__ out, int B) {
  const int lane = threadIdx.x & 31;
  const int gw = (int)((blockIdx.x * blockDim.x + threadIdx.x) >> 5);
  if (gw >= B * S) return;
  const int b = gw / S;
  const float cx = nxyz[gw * 3 + 0], cy = nxyz[gw * 3 + 1], cz = nxyz[gw * 3 + 2];
  const float* P = xyz + (size_t)b * n * 3;
  int* row = out + (size_t)gw * ns;
  int count = 0, first = -1;
  for (int base = 0; base < n; base += 32) {
    const int j = base + lane;
    bool ok = false;
    if (j < n) {
      float dx = P[j * 3 + 0] - cx;
      float dy = P[j * 3 + 1] - cy;
      float dz = P[j * 3 + 2] - cz;
      ok = (dx * dx + dy * dy + dz * dz) < r2;
    }
    unsigned mask = (unsigned)__ballot(ok);
    if (mask) {
      if (first < 0) first = base + __ffs((int)mask) - 1;
      int pre = __popc(mask & ((1u << lane) - 1u));
      if (ok && count + pre < ns) row[count + pre] = j;
      count += __popc(mask);
      if (count >= ns) break;
    }
  }
  int start = count < ns ? count : ns;
  int fill = first < 0 ? 0 : first;
  for (int k2 = start + lane; k2 < ns; k2 += 32) row[k2] = fill;
}

// ---------------------------------------------------------------------------
// SA grouped MLP: one block per (b,s) group. Gathers ns neighbors, normalizes
// coords, concats feats into LDS f16, runs 3 WMMA layers, max-pools over ns.
// ---------------------------------------------------------------------------
__global__ void k_sa_mlp(const float* __restrict__ src_xyz,
                         const float* __restrict__ src_feats, int n_src, int Cf,
                         const float* __restrict__ new_xyz,
                         const int* __restrict__ ball_idx,
                         int S, int ns, float inv_r,
                         int Cin_pad, int C1, int C2, int C3,
                         const f16* __restrict__ W0, const float* __restrict__ b0,
                         const f16* __restrict__ W1, const float* __restrict__ b1,
                         const f16* __restrict__ W2, const float* __restrict__ b2,
                         float* __restrict__ out_f, int ldA, int ldB) {
  extern __shared__ char smem[];
  f16* shA = (f16*)smem;
  f16* shB = shA + (size_t)ns * ldA;
  const int tid = threadIdx.x;
  const int blk = blockIdx.x;
  const int b = blk / S;
  const int* idxrow = ball_idx + (size_t)blk * ns;
  const float cx = new_xyz[blk * 3 + 0];
  const float cy = new_xyz[blk * 3 + 1];
  const float cz = new_xyz[blk * 3 + 2];
  for (int e = tid; e < ns * Cin_pad; e += blockDim.x) {
    int j = e / Cin_pad, k = e - j * Cin_pad;
    int id = idxrow[j];
    float v = 0.f;
    if (k < 3) {
      float p = src_xyz[((size_t)b * n_src + id) * 3 + k];
      float c = (k == 0) ? cx : ((k == 1) ? cy : cz);
      v = (p - c) * inv_r;
    } else if (k < 3 + Cf) {
      v = src_feats[((size_t)b * n_src + id) * Cf + (k - 3)];
    }
    shA[(size_t)j * Cin_pad + k] = (f16)v;
  }
  __syncthreads();
  wmma_mlp_layer(shA, Cin_pad, shB, C1, W0, b0, ns, Cin_pad, C1, tid, blockDim.x);
  __syncthreads();
  wmma_mlp_layer(shB, C1, shA, C2, W1, b1, ns, C1, C2, tid, blockDim.x);
  __syncthreads();
  wmma_mlp_layer(shA, C2, shB, C3, W2, b2, ns, C2, C3, tid, blockDim.x);
  __syncthreads();
  for (int c = tid; c < C3; c += blockDim.x) {
    float m = (float)shB[c];
    for (int j = 1; j < ns; ++j) {
      float x = (float)shB[(size_t)j * C3 + c];
      m = x > m ? x : m;
    }
    out_f[(size_t)blk * C3 + c] = m;
  }
  (void)ldB;
}

// ---------------------------------------------------------------------------
// FP interpolation: one thread per unknown point. 3-NN over m knowns (<=256),
// inverse-distance weights, writes [interp(256) | unk_feats(Cu)] as f16.
// ---------------------------------------------------------------------------
__global__ void k_fp_interp(const float* __restrict__ ux, const float* __restrict__ kx,
                            const float* __restrict__ uf, int Cu,
                            const float* __restrict__ kf,
                            int n, int m, f16* __restrict__ concat, int Cin_pad, int B) {
  int t = blockIdx.x * blockDim.x + threadIdx.x;
  if (t >= B * n) return;
  int b = t / n;
  float px = ux[t * 3 + 0], py = ux[t * 3 + 1], pz = ux[t * 3 + 2];
  float d0 = 1e30f, d1 = 1e30f, d2 = 1e30f;
  int i0 = 0, i1 = 0, i2 = 0;
  const float* K = kx + (size_t)b * m * 3;
  for (int j = 0; j < m; ++j) {
    float dx = K[j * 3 + 0] - px;
    float dy = K[j * 3 + 1] - py;
    float dz = K[j * 3 + 2] - pz;
    float dd = dx * dx + dy * dy + dz * dz;
    if (dd < d0)      { d2 = d1; i2 = i1; d1 = d0; i1 = i0; d0 = dd; i0 = j; }
    else if (dd < d1) { d2 = d1; i2 = i1; d1 = dd; i1 = j; }
    else if (dd < d2) { d2 = dd; i2 = j; }
  }
  float w0 = 1.f / (d0 + 1e-8f), w1 = 1.f / (d1 + 1e-8f), w2 = 1.f / (d2 + 1e-8f);
  float wsum = w0 + w1 + w2;
  w0 /= wsum; w1 /= wsum; w2 /= wsum;
  const float* f0 = kf + ((size_t)b * m + i0) * 256;
  const float* f1 = kf + ((size_t)b * m + i1) * 256;
  const float* f2 = kf + ((size_t)b * m + i2) * 256;
  f16* row = concat + (size_t)t * Cin_pad;
  for (int c = 0; c < 256; ++c)
    row[c] = (f16)(w0 * f0[c] + w1 * f1[c] + w2 * f2[c]);
  const float* u = uf + (size_t)t * Cu;
  for (int c = 0; c < Cu; ++c) row[256 + c] = (f16)u[c];
}

// ---------------------------------------------------------------------------
// FP MLP: 32-row tiles, 2 WMMA layers. Input staging uses the CDNA5 Tensor
// Data Mover (tensor_load_to_lds, TENSORcnt; 6-arg builtin on this
// toolchain), else async global->LDS (ASYNCcnt) via inline asm. The staged
// tile is a contiguous 1-D run of ROWS*Cin_pad f16 elements.
// Optionally writes (B,256,n) transposed.
// ---------------------------------------------------------------------------
__global__ void k_fp_mlp(const f16* __restrict__ concat, int n, int Cin_pad,
                         const f16* __restrict__ W0, const float* __restrict__ b0,
                         const f16* __restrict__ W1, const float* __restrict__ b1,
                         float* __restrict__ out, int transpose, int ldA) {
  const int ROWS = 32;
  extern __shared__ char smem[];
  f16* shA = (f16*)smem;
  f16* shB = shA + (size_t)ROWS * ldA;
  const int tid = threadIdx.x;
  const int per = n / ROWS;
  const int b = blockIdx.x / per;
  const int row0 = (blockIdx.x - b * per) * ROWS;
  const f16* gsrc = concat + ((size_t)b * n + row0) * Cin_pad;
  const unsigned lds_base = (unsigned)(size_t)(void*)shA;  // low 32b = LDS byte addr
  const unsigned nelem = (unsigned)(ROWS * Cin_pad);       // f16 elements, <= 16384

#if __has_builtin(__builtin_amdgcn_tensor_load_to_lds)
  if (tid == 0) {
    const unsigned long long ga = (unsigned long long)(const void*)gsrc;
    u32x4 g0;
    g0[0] = 0x1u;                                        // count=1, user mode
    g0[1] = lds_base;                                    // lds_addr (bytes)
    g0[2] = (unsigned)(ga & 0xFFFFFFFFu);                // global_addr[31:0]
    g0[3] = (unsigned)((ga >> 32) & 0x01FFFFFFu)         // global_addr[56:32]
          | (2u << 30);                                  // type = 2 ("image")
    i32x8 g1;
    g1[0] = (int)(1u << 16);                             // wg_mask=0, data_size=1 (2B)
    g1[1] = (int)((nelem & 0xFFFFu) << 16);              // tensor_dim0[15:0]
    g1[2] = (int)((nelem >> 16) & 0xFFFFu);              // tensor_dim0[31:16], dim1=0
    g1[3] = (int)((nelem & 0xFFFFu) << 16);              // tile_dim0 = whole run
    g1[4] = 0;                                           // tile_dim1/2 unused
    g1[5] = (int)nelem;                                  // tensor_dim0_stride[31:0]
    g1[6] = 0;
    g1[7] = 0;
    i32x4 gz4; gz4[0] = 0; gz4[1] = 0; gz4[2] = 0; gz4[3] = 0;
    i32x8 gz8;
#pragma unroll
    for (int i = 0; i < 8; ++i) gz8[i] = 0;
    __builtin_amdgcn_tensor_load_to_lds(g0, g1, gz4, gz4, gz8, 0);
    __builtin_amdgcn_s_wait_tensorcnt(0);
  }
  __syncthreads();
#else
  {
    const int total_bytes = (int)nelem * 2;
    for (int o16 = tid * 16; o16 < total_bytes; o16 += (int)blockDim.x * 16) {
      unsigned lds_addr = lds_base + (unsigned)o16;
      const void* gp = (const char*)gsrc + o16;
      asm volatile("global_load_async_to_lds_b128 %0, %1, off"
                   :: "v"(lds_addr), "v"(gp) : "memory");
    }
    asm volatile("s_wait_asynccnt 0x0" ::: "memory");
  }
  __syncthreads();
#endif

  wmma_mlp_layer(shA, Cin_pad, shB, 256, W0, b0, ROWS, Cin_pad, 256, tid, blockDim.x);
  __syncthreads();
  wmma_mlp_layer(shB, 256, shA, 256, W1, b1, ROWS, 256, 256, tid, blockDim.x);
  __syncthreads();
  for (int e = tid; e < ROWS * 256; e += blockDim.x) {
    int j = e >> 8, c = e & 255;
    float v = (float)shA[(size_t)j * 256 + c];
    if (transpose) out[((size_t)b * 256 + c) * n + (row0 + j)] = v;
    else           out[(((size_t)b * n) + row0 + j) * 256 + c] = v;
  }
}

__global__ void k_copy_f32(const float* __restrict__ s, float* __restrict__ d, int n) {
  int t = blockIdx.x * blockDim.x + threadIdx.x;
  if (t < n) d[t] = s[t];
}

// ---------------------------------------------------------------------------
// Host orchestration
// ---------------------------------------------------------------------------
extern "C" void kernel_launch(void* const* d_in, const int* in_sizes, int n_in,
                              void* d_out, int out_size, void* d_ws, size_t ws_size,
                              hipStream_t stream) {
  (void)in_sizes; (void)n_in; (void)out_size; (void)ws_size;
  const int B = 8, N = 20000;
  const float* pc = (const float*)d_in[0];

  // 18 layers: sa1(3) sa2(3) sa3(3) sa4(3) fp1(2) fp2(2) fp3(2)
  const int NL = 18;
  static const int dco[NL] = {64,64,128, 128,128,256, 128,128,256, 128,128,256,
                              256,256, 256,256, 256,256};
  static const int dci[NL] = {3,64,64, 131,128,128, 259,128,128, 259,128,128,
                              512,256, 512,256, 384,256};
  const float* Wf[NL]; const float* Bias[NL];
  for (int i = 0; i < NL; ++i) {
    Wf[i]   = (const float*)d_in[1 + 2 * i];
    Bias[i] = (const float*)d_in[2 + 2 * i];
  }

  // workspace bump allocator (256B aligned)
  char* base = (char*)d_ws;
  size_t off = 0;
  auto alloc = [&](size_t bytes) -> char* {
    char* p = base + off;
    off += (bytes + 255) & ~(size_t)255;
    return p;
  };

  f16* W16[NL]; int cinp[NL];
  for (int i = 0; i < NL; ++i) {
    cinp[i] = ((dci[i] + 31) / 32) * 32;
    W16[i] = (f16*)alloc((size_t)dco[i] * cinp[i] * sizeof(f16));
  }
  float* dists = (float*)alloc((size_t)B * N * 4);
  int* inds1 = (int*)alloc((size_t)B * 512 * 4);
  int* inds2 = (int*)alloc((size_t)B * 256 * 4);
  int* inds3 = (int*)alloc((size_t)B * 128 * 4);
  int* inds4 = (int*)alloc((size_t)B * 64 * 4);
  float* xyz1 = (float*)alloc((size_t)B * 512 * 3 * 4);
  float* xyz2 = (float*)alloc((size_t)B * 256 * 3 * 4);
  float* xyz3 = (float*)alloc((size_t)B * 128 * 3 * 4);
  float* xyz4 = (float*)alloc((size_t)B * 64 * 3 * 4);
  int* bidx1 = (int*)alloc((size_t)B * 512 * 64 * 4);
  int* bidx2 = (int*)alloc((size_t)B * 256 * 32 * 4);
  int* bidx3 = (int*)alloc((size_t)B * 128 * 16 * 4);
  int* bidx4 = (int*)alloc((size_t)B * 64 * 16 * 4);
  float* f1 = (float*)alloc((size_t)B * 512 * 128 * 4);
  float* f2 = (float*)alloc((size_t)B * 256 * 256 * 4);
  float* f3 = (float*)alloc((size_t)B * 128 * 256 * 4);
  float* f4 = (float*)alloc((size_t)B * 64 * 256 * 4);
  float* g1 = (float*)alloc((size_t)B * 128 * 256 * 4);  // fp1 out
  float* g2 = (float*)alloc((size_t)B * 256 * 256 * 4);  // fp2 out
  f16* cc1 = (f16*)alloc((size_t)B * 128 * 512 * sizeof(f16));
  f16* cc2 = (f16*)alloc((size_t)B * 256 * 512 * sizeof(f16));
  f16* cc3 = (f16*)alloc((size_t)B * 512 * 384 * sizeof(f16));

  // weight conversion (deterministic, part of every launch)
  for (int i = 0; i < NL; ++i) {
    int tot = dco[i] * cinp[i];
    k_convert_w<<<(tot + 255) / 256, 256, 0, stream>>>(Wf[i], W16[i], dco[i], dci[i], cinp[i]);
  }

  // ---- SA1: N=20000 -> S=512, ns=64, r=0.04, dims 3->64->64->128 ----
  k_fps<<<B, 256, 0, stream>>>(pc, N, 512, inds1, dists);
  k_gather_xyz<<<(B * 512 + 255) / 256, 256, 0, stream>>>(pc, inds1, xyz1, N, 512, B);
  k_ball_query<<<(B * 512 * 32 + 255) / 256, 256, 0, stream>>>(pc, xyz1, N, 512, 64,
                                                               0.0016f, bidx1, B);
  {
    int ldA = 64, ldB = 128;  // max(Cin_pad=32,C2=64), max(C1=64,C3=128)
    size_t sh = (size_t)64 * (ldA + ldB) * sizeof(f16);
    k_sa_mlp<<<B * 512, 256, sh, stream>>>(pc, nullptr, N, 0, xyz1, bidx1, 512, 64,
        25.f, 32, 64, 64, 128,
        W16[0], Bias[0], W16[1], Bias[1], W16[2], Bias[2], f1, ldA, ldB);
  }

  // ---- SA2: 512 -> 256, ns=32, r=0.1, dims 131->128->128->256 ----
  k_fps<<<B, 256, 0, stream>>>(xyz1, 512, 256, inds2, dists);
  k_gather_xyz<<<(B * 256 + 255) / 256, 256, 0, stream>>>(xyz1, inds2, xyz2, 512, 256, B);
  k_ball_query<<<(B * 256 * 32 + 255) / 256, 256, 0, stream>>>(xyz1, xyz2, 512, 256, 32,
                                                               0.01f, bidx2, B);
  {
    int ldA = 160, ldB = 256;
    size_t sh = (size_t)32 * (ldA + ldB) * sizeof(f16);
    k_sa_mlp<<<B * 256, 256, sh, stream>>>(xyz1, f1, 512, 128, xyz2, bidx2, 256, 32,
        10.f, 160, 128, 128, 256,
        W16[3], Bias[3], W16[4], Bias[4], W16[5], Bias[5], f2, ldA, ldB);
  }

  // ---- SA3: 256 -> 128, ns=16, r=0.2, dims 259->128->128->256 ----
  k_fps<<<B, 256, 0, stream>>>(xyz2, 256, 128, inds3, dists);
  k_gather_xyz<<<(B * 128 + 255) / 256, 256, 0, stream>>>(xyz2, inds3, xyz3, 256, 128, B);
  k_ball_query<<<(B * 128 * 32 + 255) / 256, 256, 0, stream>>>(xyz2, xyz3, 256, 128, 16,
                                                               0.04f, bidx3, B);
  {
    int ldA = 288, ldB = 256;
    size_t sh = (size_t)16 * (ldA + ldB) * sizeof(f16);
    k_sa_mlp<<<B * 128, 256, sh, stream>>>(xyz2, f2, 256, 256, xyz3, bidx3, 128, 16,
        5.f, 288, 128, 128, 256,
        W16[6], Bias[6], W16[7], Bias[7], W16[8], Bias[8], f3, ldA, ldB);
  }

  // ---- SA4: 128 -> 64, ns=16, r=0.3, dims 259->128->128->256 ----
  k_fps<<<B, 256, 0, stream>>>(xyz3, 128, 64, inds4, dists);
  k_gather_xyz<<<(B * 64 + 255) / 256, 256, 0, stream>>>(xyz3, inds4, xyz4, 128, 64, B);
  k_ball_query<<<(B * 64 * 32 + 255) / 256, 256, 0, stream>>>(xyz3, xyz4, 128, 64, 16,
                                                              0.09f, bidx4, B);
  {
    int ldA = 288, ldB = 256;
    size_t sh = (size_t)16 * (ldA + ldB) * sizeof(f16);
    k_sa_mlp<<<B * 64, 256, sh, stream>>>(xyz3, f3, 128, 256, xyz4, bidx4, 64, 16,
        1.f / 0.3f, 288, 128, 128, 256,
        W16[9], Bias[9], W16[10], Bias[10], W16[11], Bias[11], f4, ldA, ldB);
  }

  // ---- FP1: sa4(64) -> sa3(128), Cin=512 ----
  k_fp_interp<<<(B * 128 + 255) / 256, 256, 0, stream>>>(xyz3, xyz4, f3, 256, f4,
                                                         128, 64, cc1, 512, B);
  {
    size_t sh = (size_t)32 * (512 + 256) * sizeof(f16);
    k_fp_mlp<<<B * (128 / 32), 256, sh, stream>>>(cc1, 128, 512,
        W16[12], Bias[12], W16[13], Bias[13], g1, 0, 512);
  }

  // ---- FP2: sa3(128) -> sa2(256), Cin=512 ----
  k_fp_interp<<<(B * 256 + 255) / 256, 256, 0, stream>>>(xyz2, xyz3, f2, 256, g1,
                                                         256, 128, cc2, 512, B);
  {
    size_t sh = (size_t)32 * (512 + 256) * sizeof(f16);
    k_fp_mlp<<<B * (256 / 32), 256, sh, stream>>>(cc2, 256, 512,
        W16[14], Bias[14], W16[15], Bias[15], g2, 0, 512);
  }

  // ---- FP3: sa2(256) -> sa1(512), Cin=384; write transposed (B,256,512) ----
  k_fp_interp<<<(B * 512 + 255) / 256, 256, 0, stream>>>(xyz1, xyz2, f1, 128, g2,
                                                         512, 256, cc3, 384, B);
  {
    size_t sh = (size_t)32 * (384 + 256) * sizeof(f16);
    k_fp_mlp<<<B * (512 / 32), 256, sh, stream>>>(cc3, 512, 384,
        W16[16], Bias[16], W16[17], Bias[17], (float*)d_out, 1, 384);
  }

  // ---- fp3_xyz = sa1_xyz appended after features ----
  k_copy_f32<<<(B * 512 * 3 + 255) / 256, 256, 0, stream>>>(
      xyz1, (float*)d_out + (size_t)B * 256 * 512, B * 512 * 3);
}
